// Experts_3848290697280
// MI455X (gfx1250) — matrix operand order repaired and checked by reading, not compiled, not run on hardware
//
#include <hip/hip_runtime.h>
#include <math.h>
#include <stdint.h>

// ---------------------------------------------------------------------------
// Fused grouped-expert MLP:  out = gelu(X @ W1 + b1) @ W2 + b2   (per expert)
// E=8, C=16384, D=512, F=2048.  ~950 FLOP/byte -> compute bound -> bf16 WMMA.
// Pass 1 (once): transpose W1/W2 into bf16 [K-contiguous-per-lane] layouts in
// workspace. Pass 2: fused 2-GEMM kernel; weight chunks double-buffered in
// LDS via gfx1250 async b128 global->LDS loads (ASYNCcnt pipelined); fragment
// loads software-pipelined one WMMA ahead so ds waits are partial-count.
// ---------------------------------------------------------------------------

typedef __attribute__((ext_vector_type(16))) __bf16 v16bf;
typedef __attribute__((ext_vector_type(8)))  float  v8f;
typedef __attribute__((ext_vector_type(4)))  unsigned int u32x4;
typedef __attribute__((ext_vector_type(4)))  float  f32x4;

#define E_ 8
#define C_ 16384
#define D_ 512
#define F_ 2048

#define ROWS 128           // C-rows per workgroup tile (16 waves / 512 thr)
#define FCH  32            // F-chunk per iteration
#define XP   (D_ + 8)      // Xs / W1Ts pitch in bf16 elems (rows 16B aligned)
#define HP   (FCH + 8)     // Hs pitch
#define WP   (FCH + 8)     // W2Ts pitch

#define USE_ASYNC_ASM 1

union Frag { v16bf v; u32x4 q[2]; };

__device__ __forceinline__ float gelu_tanh(float x) {
    float u = 0.7978845608028654f * (x + 0.044715f * x * x * x);
    return 0.5f * x * (1.0f + tanhf(u));
}

// global->LDS async b128 copy (gfx1250). LDS offset = low 32 bits of generic
// address (ISA: LDS aperture uses addr[31:0]).
__device__ __forceinline__ void async_copy_b128(const __bf16* gsrc, __bf16* ldst) {
#if USE_ASYNC_ASM
    unsigned loff = (unsigned)(uintptr_t)ldst;
    asm volatile("global_load_async_to_lds_b128 %0, %1, off"
                 :: "v"(loff), "v"(gsrc) : "memory");
#else
    *(u32x4*)ldst = *(const u32x4*)gsrc;
#endif
}
__device__ __forceinline__ void wait_async_all() {
#if USE_ASYNC_ASM
    asm volatile("s_wait_asynccnt 0x0" ::: "memory");
#endif
}
// keep the prefetched next chunk (8 ops/thread) in flight
__device__ __forceinline__ void wait_async_keep8() {
#if USE_ASYNC_ASM
    asm volatile("s_wait_asynccnt 0x8" ::: "memory");
#endif
}

__device__ __forceinline__ void load_frag(Frag& f, const __bf16* p0, const __bf16* p1) {
    f.q[0] = *(const u32x4*)p0;
    f.q[1] = *(const u32x4*)p1;
}

// ---------------------------------------------------------------------------
// Pass 1: tiled transpose f32 -> bf16.  src: [E][R][Cc] f32, dst: [E][Cc][R].
// ---------------------------------------------------------------------------
__global__ __launch_bounds__(256)
void transpose_to_bf16(const float* __restrict__ src, __bf16* __restrict__ dst,
                       int R, int Cc) {
    __shared__ float tile[32][33];
    const int e  = blockIdx.z;
    const int c0 = blockIdx.x * 32;
    const int r0 = blockIdx.y * 32;
    const int tx = threadIdx.x, ty = threadIdx.y;        // 32 x 8
    const float* s = src + (size_t)e * R * Cc;
    __bf16*      d = dst + (size_t)e * Cc * R;
#pragma unroll
    for (int i = 0; i < 4; ++i)
        tile[ty + i * 8][tx] = s[(size_t)(r0 + ty + i * 8) * Cc + (c0 + tx)];
    __syncthreads();
#pragma unroll
    for (int i = 0; i < 4; ++i)
        d[(size_t)(c0 + ty + i * 8) * R + (r0 + tx)] = (__bf16)tile[tx][ty + i * 8];
}

// ---------------------------------------------------------------------------
// Pass 2: fused expert MLP.  PRETRANS: weights double-buffer staged from bf16
// transposed workspace via async b128 copies; else transposed in-kernel.
// ---------------------------------------------------------------------------
template<bool PRETRANS>
__global__ __launch_bounds__(512)
void moe_ffn_fused(const float* __restrict__ X,  const float* __restrict__ W1,
                   const float* __restrict__ b1, const float* __restrict__ W2,
                   const float* __restrict__ b2, float* __restrict__ Out,
                   const __bf16* __restrict__ W1T_g, const __bf16* __restrict__ W2T_g)
{
    constexpr int NBUF = PRETRANS ? 2 : 1;
    __shared__ alignas(16) __bf16 Xs  [ROWS * XP];          // X tile bf16 [r][k]
    __shared__ alignas(16) __bf16 W1Ts[NBUF * FCH * XP];    // W1 chunk   [f][k]
    __shared__ alignas(16) __bf16 Hs  [ROWS * HP];          // gelu(h)    [r][f]
    __shared__ alignas(16) __bf16 W2Ts[NBUF * D_ * WP];     // W2 chunk   [d][kf]

    const int tid  = threadIdx.x;
    const int wave = tid >> 5;
    const int lane = tid & 31;
    const int lm   = lane & 15;
    const int lh   = lane >> 4;

    const int tilesPerExpert = C_ / ROWS;                // 128
    const int e       = blockIdx.x / tilesPerExpert;
    const int rowBase = (blockIdx.x % tilesPerExpert) * ROWS;

    const float* Xg  = X  + ((size_t)e * C_ + rowBase) * D_;
    float*       Og  = Out + ((size_t)e * C_ + rowBase) * D_;
    const float* W1g = W1 + (size_t)e * D_ * F_;
    const float* W2g = W2 + (size_t)e * F_ * D_;
    const __bf16* W1Tg = W1T_g + (size_t)e * F_ * D_;
    const __bf16* W2Tg = W2T_g + (size_t)e * D_ * F_;

    // issue one f-chunk of weight staging (8 async b128 ops per thread)
    auto stage_weights = [&](int fc, __bf16* w1buf, __bf16* w2buf) {
        int seg = tid & 63;                // 16B segment within 512-elem row
        int r0  = tid >> 6;                // 0..7
#pragma unroll
        for (int i = 0; i < 4; ++i) {
            int f = r0 + i * 8;
            async_copy_b128(W1Tg + (size_t)(fc + f) * D_ + seg * 8,
                            w1buf + f * XP + seg * 8);
        }
        int quad = tid & 3;
        int r1   = tid >> 2;               // 0..127
#pragma unroll
        for (int i = 0; i < 4; ++i) {
            int d = r1 + i * 128;
            async_copy_b128(W2Tg + (size_t)d * F_ + fc + quad * 8,
                            w2buf + d * WP + quad * 8);
        }
    };

    // prologue: prefetch chunk 0 into buffer 0 before X staging
    if constexpr (PRETRANS) stage_weights(0, W1Ts, W2Ts);

    // ---- stage X tile once (f32 -> bf16) ----
    for (int i = 0; i < (ROWS * D_) / (512 * 4); ++i) {   // 32 iters
        int fidx = (tid + i * 512) * 4;
        int r = fidx / D_, k = fidx % D_;
        f32x4 v = *(const f32x4*)(Xg + (size_t)r * D_ + k);
        __bf16* dst = Xs + r * XP + k;
        dst[0] = (__bf16)v.x; dst[1] = (__bf16)v.y;
        dst[2] = (__bf16)v.z; dst[3] = (__bf16)v.w;
    }

    // stage-1: h is 128x32 -> 8M x 2N tiles, one per wave
    const int mt1 = wave >> 1;            // 0..7
    const int nt1 = wave & 1;
    // stage-2 / output: 8M x 32N tiles, 16 per wave
    const int mt2  = wave >> 1;
    const int cbas = (wave & 1) * 256;

    v8f cacc[16];
#pragma unroll
    for (int t = 0; t < 16; ++t) cacc[t] = {};

    __syncthreads();

    for (int fc = 0; fc < F_; fc += FCH) {
        const int cb = (fc / FCH) & (NBUF - 1);
        __bf16* w1c = W1Ts + cb * (FCH * XP);
        __bf16* w2c = W2Ts + cb * (D_ * WP);

        if constexpr (PRETRANS) {
            // pipeline: issue next chunk, then retire current chunk's 8 ops
            if (fc + FCH < F_) {
                const int nb = cb ^ 1;
                stage_weights(fc + FCH, W1Ts + nb * (FCH * XP), W2Ts + nb * (D_ * WP));
                wait_async_keep8();
            } else {
                wait_async_all();
            }
        } else {
            // in-kernel transpose from f32 weights
            {
                int f = tid & 31;
                for (int k = tid >> 5; k < D_; k += 16)
                    w1c[f * XP + k] = (__bf16)W1g[(size_t)k * F_ + (fc + f)];
            }
            {
                int d4 = (tid & 127) * 4;
                for (int kf = tid >> 7; kf < FCH; kf += 4) {
                    f32x4 v = *(const f32x4*)(W2g + (size_t)(fc + kf) * D_ + d4);
                    w2c[(d4 + 0) * WP + kf] = (__bf16)v.x;
                    w2c[(d4 + 1) * WP + kf] = (__bf16)v.y;
                    w2c[(d4 + 2) * WP + kf] = (__bf16)v.z;
                    w2c[(d4 + 3) * WP + kf] = (__bf16)v.w;
                }
            }
        }
        __syncthreads();

        // ---- stage 1: h_tile = X_tile @ W1_chunk + b1 (K = 512) ----
        // fragment loads pipelined one K-step ahead
        v8f hacc;
        {
            float bias = b1[(size_t)e * F_ + fc + nt1 * 16 + lm];
#pragma unroll
            for (int j = 0; j < 8; ++j) hacc[j] = bias;
        }
        const __bf16* arow = Xs  + (mt1 * 16 + lm) * XP + lh * 8;
        const __bf16* brow = w1c + (nt1 * 16 + lm) * XP + lh * 8;
        Frag a1[2], b1f[2];
        load_frag(a1[0], arow, arow + 16);
        load_frag(b1f[0], brow, brow + 16);
#pragma unroll
        for (int kk = 0; kk < D_; kk += 32) {
            const int cur = (kk >> 5) & 1;
            if (kk + 32 < D_) {
                const int nxt = cur ^ 1;
                load_frag(a1[nxt], arow + (kk + 32), arow + (kk + 32) + 16);
                load_frag(b1f[nxt], brow + (kk + 32), brow + (kk + 32) + 16);
            }
            hacc = __builtin_amdgcn_wmma_f32_16x16x32_bf16(
                       false, a1[cur].v, false, b1f[cur].v, (short)0, hacc, false, false);
        }
        // GELU + bf16 h to LDS (C/D layout: VGPR j -> M = j + 8*lh)
#pragma unroll
        for (int j = 0; j < 8; ++j) {
            float g = gelu_tanh(hacc[j]);
            Hs[(mt1 * 16 + lh * 8 + j) * HP + nt1 * 16 + lm] = (__bf16)g;
        }
        __syncthreads();

        // ---- stage 2: out += h @ W2_chunk (K = 32, one WMMA step each) ----
        Frag a2;
        const __bf16* ap2 = Hs + (mt2 * 16 + lm) * HP + lh * 8;
        load_frag(a2, ap2, ap2 + 16);
        const __bf16* brow2 = w2c + (cbas + lm) * WP + lh * 8;
        Frag b2f[2];
        load_frag(b2f[0], brow2, brow2 + 16);
#pragma unroll
        for (int nt = 0; nt < 16; ++nt) {
            const int cur = nt & 1;
            if (nt + 1 < 16) {
                const __bf16* bp = brow2 + (nt + 1) * (16 * WP);
                load_frag(b2f[cur ^ 1], bp, bp + 16);
            }
            cacc[nt] = __builtin_amdgcn_wmma_f32_16x16x32_bf16(
                           false, a2.v, false, b2f[cur].v, (short)0, cacc[nt], false, false);
        }
        __syncthreads();   // releases Hs and the buffer staged next iteration
    }

    // ---- epilogue: add b2, write f32 ----
#pragma unroll
    for (int nt = 0; nt < 16; ++nt) {
        int col = cbas + nt * 16 + lm;
        float bias = b2[(size_t)e * D_ + col];
#pragma unroll
        for (int j = 0; j < 8; ++j) {
            int row = mt2 * 16 + lh * 8 + j;
            Og[(size_t)row * D_ + col] = cacc[nt][j] + bias;
        }
    }
}

extern "C" void kernel_launch(void* const* d_in, const int* in_sizes, int n_in,
                              void* d_out, int out_size, void* d_ws, size_t ws_size,
                              hipStream_t stream) {
    const float* X  = (const float*)d_in[0];
    const float* W1 = (const float*)d_in[1];
    const float* b1 = (const float*)d_in[2];
    const float* W2 = (const float*)d_in[3];
    const float* b2 = (const float*)d_in[4];
    float* Out = (float*)d_out;

    const size_t elemsW = (size_t)E_ * D_ * F_;          // 8 Mi elems each
    const size_t needWs = 2 * elemsW * 2;                // 32 MiB bf16

    dim3 block(512);
    dim3 grid(E_ * (C_ / ROWS));                         // 1024 workgroups

    if (ws_size >= needWs) {
        __bf16* W1T_g = (__bf16*)d_ws;                   // [E][F][D]
        __bf16* W2T_g = (__bf16*)((char*)d_ws + elemsW * 2); // [E][D][F]
        // Pass 1: one-time weight transposes (f32 -> bf16)
        transpose_to_bf16<<<dim3(F_ / 32, D_ / 32, E_), dim3(32, 8), 0, stream>>>(
            W1, W1T_g, D_, F_);
        transpose_to_bf16<<<dim3(D_ / 32, F_ / 32, E_), dim3(32, 8), 0, stream>>>(
            W2, W2T_g, F_, D_);
        // Pass 2: fused MLP with pipelined async weight staging
        moe_ffn_fused<true><<<grid, block, 0, stream>>>(
            X, W1, b1, W2, b2, Out, W1T_g, W2T_g);
    } else {
        moe_ffn_fused<false><<<grid, block, 0, stream>>>(
            X, W1, b1, W2, b2, Out, (const __bf16*)nullptr, (const __bf16*)nullptr);
    }
}
